// MultiHeadAttention_38268158607584
// MI455X (gfx1250) — matrix operand order repaired and checked by reading, not compiled, not run on hardware
//
#include <hip/hip_runtime.h>
#include <hip/hip_bf16.h>

// ---------------------------------------------------------------------------
// MI455X (gfx1250, wave32) transformer attention block.
//   proj(Q/K/V f32->f16, WMMA) -> flash-style attention (WMMA QK^T, online
//   softmax with lane-local stats + single end butterfly, attn written once,
//   WMMA P*V via V^T) -> out-proj + residual (WMMA) -> in-place LayerNorm.
// All matrix math uses v_wmma_f32_16x16x32_f16 (f16 in, f32 accumulate).
// ---------------------------------------------------------------------------

typedef __attribute__((ext_vector_type(16))) _Float16 v16h;
typedef __attribute__((ext_vector_type(8)))  _Float16 v8h;
typedef __attribute__((ext_vector_type(8)))  float    v8f;

#define S_LEN   2048
#define D_MODEL 1024
#define N_HEAD  16
#define D_K     64
#define BATCH   4
#define ROWS    (BATCH * S_LEN)          // 8192
#define OUT_ELEMS ((size_t)ROWS * D_MODEL)

__device__ __forceinline__ v8f wmma_f16(v16h a, v16h b, v8f c) {
  return __builtin_amdgcn_wmma_f32_16x16x32_f16(false, a, false, b,
                                                (short)0, c, false, false);
}

__device__ __forceinline__ v16h cat8(v8h lo, v8h hi) {
  return __builtin_shufflevector(lo, hi, 0,1,2,3,4,5,6,7,8,9,10,11,12,13,14,15);
}

// A fragment (16x32, f16) from row-major f32, row stride ld.
// lane<16: m=m0+lane,   K in {k0..k0+15}
// lane>=16:m=m0+lane-16,K in {k0+16..k0+31}
__device__ __forceinline__ v16h a_frag_f32(const float* __restrict__ X, int ld,
                                           int m0, int k0, int lane) {
  const int half = lane >> 4, mr = lane & 15;
  const float* p = X + (size_t)(m0 + mr) * ld + k0 + half * 16;
  float4 x0 = *(const float4*)(p + 0);
  float4 x1 = *(const float4*)(p + 4);
  float4 x2 = *(const float4*)(p + 8);
  float4 x3 = *(const float4*)(p + 12);
  v16h a;
  a[0]=(_Float16)x0.x; a[1]=(_Float16)x0.y; a[2]=(_Float16)x0.z; a[3]=(_Float16)x0.w;
  a[4]=(_Float16)x1.x; a[5]=(_Float16)x1.y; a[6]=(_Float16)x1.z; a[7]=(_Float16)x1.w;
  a[8]=(_Float16)x2.x; a[9]=(_Float16)x2.y; a[10]=(_Float16)x2.z; a[11]=(_Float16)x2.w;
  a[12]=(_Float16)x3.x; a[13]=(_Float16)x3.y; a[14]=(_Float16)x3.z; a[15]=(_Float16)x3.w;
  return a;
}

// B fragment (32x16, f16) where B[k][n] = W[n0+n][k0+k] (row-major W, stride ld).
// lane<16: column n=lane, K=0..15 -> W[n][k0..k0+15]; lane>=16: K=16..31.
__device__ __forceinline__ v16h b_frag_f32(const float* __restrict__ W, int ld,
                                           int n0, int k0, int lane) {
  const float* p = W + (size_t)(n0 + (lane & 15)) * ld + k0 + (lane >> 4) * 16;
  float4 x0 = *(const float4*)(p + 0);
  float4 x1 = *(const float4*)(p + 4);
  float4 x2 = *(const float4*)(p + 8);
  float4 x3 = *(const float4*)(p + 12);
  v16h b;
  b[0]=(_Float16)x0.x; b[1]=(_Float16)x0.y; b[2]=(_Float16)x0.z; b[3]=(_Float16)x0.w;
  b[4]=(_Float16)x1.x; b[5]=(_Float16)x1.y; b[6]=(_Float16)x1.z; b[7]=(_Float16)x1.w;
  b[8]=(_Float16)x2.x; b[9]=(_Float16)x2.y; b[10]=(_Float16)x2.z; b[11]=(_Float16)x2.w;
  b[12]=(_Float16)x3.x; b[13]=(_Float16)x3.y; b[14]=(_Float16)x3.z; b[15]=(_Float16)x3.w;
  return b;
}

__device__ __forceinline__ v16h a_frag_f16(const _Float16* __restrict__ X, int ld,
                                           int m0, int k0, int lane) {
  const _Float16* p = X + (size_t)(m0 + (lane & 15)) * ld + k0 + (lane >> 4) * 16;
  v8h lo = *(const v8h*)(p);
  v8h hi = *(const v8h*)(p + 8);
  return cat8(lo, hi);
}

__device__ __forceinline__ v16h b_frag_f16(const _Float16* __restrict__ W, int ld,
                                           int n0, int k0, int lane) {
  const _Float16* p = W + (size_t)(n0 + (lane & 15)) * ld + k0 + (lane >> 4) * 16;
  v8h lo = *(const v8h*)(p);
  v8h hi = *(const v8h*)(p + 8);
  return cat8(lo, hi);
}

// ---------------------------------------------------------------------------
// Kernel 1: input projection  Y = X @ W^T  (f32 -> f16, head-major output).
// MODE 0: Q, scaled by 1/sqrt(d_k), layout [b][h][s][d]
// MODE 1: K,                        layout [b][h][s][d]
// MODE 2: V,                        layout [b][h][d][s]  (transposed)
// Each wave computes a 64(M) x 16(N) output tile; grid covers 8192x1024.
// ---------------------------------------------------------------------------
template <int MODE>
__global__ void proj_kernel(const float* __restrict__ X,
                            const float* __restrict__ W,
                            _Float16* __restrict__ Y) {
  const int lane = threadIdx.x & 31;
  const int wave = blockIdx.x * (blockDim.x >> 5) + (threadIdx.x >> 5);
  const int nt = wave & 63;       // 64 N-tiles of 16
  const int mt = wave >> 6;       // 128 M-tiles of 64
  const int m0 = mt * 64, n0 = nt * 16;

  v8f zero = {0.f,0.f,0.f,0.f,0.f,0.f,0.f,0.f};
  v8f acc[4] = {zero, zero, zero, zero};

  for (int k0 = 0; k0 < D_MODEL; k0 += 32) {
    v16h b = b_frag_f32(W, D_MODEL, n0, k0, lane);
#pragma unroll
    for (int i = 0; i < 4; ++i) {
      v16h a = a_frag_f32(X, D_MODEL, m0 + 16 * i, k0, lane);
      acc[i] = wmma_f16(a, b, acc[i]);
    }
  }

  const int half = lane >> 4, nl = lane & 15;
  const int n = n0 + nl;
  const int h = n >> 6, d = n & 63;
#pragma unroll
  for (int i = 0; i < 4; ++i) {
#pragma unroll
    for (int r = 0; r < 8; ++r) {
      const int m = m0 + i * 16 + r + 8 * half;
      const int bb = m >> 11, s = m & (S_LEN - 1);
      float v = acc[i][r];
      if (MODE == 0) v *= 0.125f;   // 1/sqrt(64)
      size_t idx;
      if (MODE == 2)
        idx = (((size_t)(bb * N_HEAD + h) * D_K + d) * S_LEN) + s;
      else
        idx = (((size_t)(bb * N_HEAD + h) * S_LEN + s) * D_K) + d;
      Y[idx] = (_Float16)v;
    }
  }
}

// ---------------------------------------------------------------------------
// Kernel 2: attention. One wave per (b,h,16-row q block). Two phases:
//  phase 1: scores = Q K^T (+mask); LANE-LOCAL online (max,sum) per row —
//           no cross-lane traffic in the loop; one 4-step butterfly at end.
//  phase 2: recompute scores, p = exp(s-m)/l, write attn once, stage P^T in
//           LDS, out^T += V^T @ P^T via WMMA.
// ---------------------------------------------------------------------------
__global__ void attn_kernel(const _Float16* __restrict__ Qh,
                            const _Float16* __restrict__ Kh,
                            const _Float16* __restrict__ VhT,
                            const int* __restrict__ mask,
                            float* __restrict__ attn,
                            float* __restrict__ AO) {
  __shared__ _Float16 pbuf[8][16][32];   // per-wave P tile (16q x 32k), 8 KB

  const int lane = threadIdx.x & 31;
  const int w    = threadIdx.x >> 5;
  const int z    = blockIdx.x;           // 0..1023
  const int bh   = z >> 4;               // 0..63
  const int qc   = z & 15;
  const int b    = bh >> 4;
  const int h    = bh & 15;
  const int q0   = qc * 128 + w * 16;

  const _Float16* Q  = Qh  + (size_t)bh * S_LEN * D_K;
  const _Float16* K  = Kh  + (size_t)bh * S_LEN * D_K;
  const _Float16* VT = VhT + (size_t)bh * D_K * S_LEN;
  const int* mrow    = mask + (size_t)b * S_LEN * S_LEN + (size_t)q0 * S_LEN;
  float* attn_bh     = attn + (size_t)bh * S_LEN * S_LEN + (size_t)q0 * S_LEN;

  const int half = lane >> 4, nl = lane & 15;
  const v8f zero = {0.f,0.f,0.f,0.f,0.f,0.f,0.f,0.f};

  // Q fragments for this 16-row block (K-dim = 64 -> two 16x32 A frags)
  const v16h aq0 = a_frag_f16(Q, D_K, q0, 0,  lane);
  const v16h aq1 = a_frag_f16(Q, D_K, q0, 32, lane);

  // ---- Phase 1: lane-local online softmax statistics ----
  float mloc[8], lloc[8];
#pragma unroll
  for (int r = 0; r < 8; ++r) { mloc[r] = -3.0e38f; lloc[r] = 0.f; }

  for (int kt = 0; kt < S_LEN / 16; ++kt) {
    const int kcol0 = kt * 16;
    v16h b0 = b_frag_f16(K, D_K, kcol0, 0,  lane);
    v16h b1 = b_frag_f16(K, D_K, kcol0, 32, lane);
    // prefetch next K tile (-> global_prefetch_b8)
    __builtin_prefetch(K + (size_t)(kcol0 + 16 + nl) * D_K + (half << 4), 0, 1);
    v8f c = zero;
    c = wmma_f16(aq0, b0, c);
    c = wmma_f16(aq1, b1, c);
    const int keyc = kcol0 + nl;
#pragma unroll
    for (int r = 0; r < 8; ++r) {
      const int q = r + 8 * half;
      const int mk = mrow[(size_t)q * S_LEN + keyc];
      const float s = mk ? c[r] : -1.0e9f;
      const float mnew = fmaxf(mloc[r], s);
      lloc[r] = lloc[r] * __expf(mloc[r] - mnew) + __expf(s - mnew);
      mloc[r] = mnew;
    }
  }

  // Single cross-lane (m,l) pair-merge butterfly over each 16-lane group.
  float mrun[8], linv[8];
#pragma unroll
  for (int r = 0; r < 8; ++r) {
    float m = mloc[r], l = lloc[r];
#pragma unroll
    for (int o = 1; o < 16; o <<= 1) {
      const float mo = __shfl_xor(m, o, 32);
      const float lo = __shfl_xor(l, o, 32);
      const float mn = fmaxf(m, mo);
      l = l * __expf(m - mn) + lo * __expf(mo - mn);
      m = mn;
    }
    mrun[r] = m;
    linv[r] = 1.0f / l;
  }

  // ---- Phase 2: final probabilities + P*V ----
  v8f acc[4] = {zero, zero, zero, zero};   // out^T: 64(d) x 16(q)
  for (int kt2 = 0; kt2 < S_LEN / 32; ++kt2) {
#pragma unroll
    for (int sub = 0; sub < 2; ++sub) {
      const int kcol0 = kt2 * 32 + sub * 16;
      v16h b0 = b_frag_f16(K, D_K, kcol0, 0,  lane);
      v16h b1 = b_frag_f16(K, D_K, kcol0, 32, lane);
      __builtin_prefetch(K + (size_t)(kcol0 + 16 + nl) * D_K + (half << 4), 0, 1);
      v8f c = zero;
      c = wmma_f16(aq0, b0, c);
      c = wmma_f16(aq1, b1, c);
      const int keyc = kcol0 + nl;
#pragma unroll
      for (int r = 0; r < 8; ++r) {
        const int q = r + 8 * half;
        const int mk = mrow[(size_t)q * S_LEN + keyc];
        const float s = mk ? c[r] : -1.0e9f;
        const float p = __expf(s - mrun[r]) * linv[r];
        attn_bh[(size_t)q * S_LEN + keyc] = p;                 // written once
        pbuf[w][q][sub * 16 + nl] = (_Float16)p;               // stage P^T
      }
    }
    // Same-wave LDS RAW: DS ops are in order; compiler inserts s_wait_dscnt.
    // B fragment of P^T (32 keys x 16 q): lane reads row q=nl, K-half by lane.
    v8h plo = *(const v8h*)&pbuf[w][nl][16 * half];
    v8h phi = *(const v8h*)&pbuf[w][nl][16 * half + 8];
    v16h bp = cat8(plo, phi);
#pragma unroll
    for (int ds = 0; ds < 4; ++ds) {
      v16h av = a_frag_f16(VT, S_LEN, ds * 16, kt2 * 32, lane);  // V^T tile
      acc[ds] = wmma_f16(av, bp, acc[ds]);
    }
  }

  // Store out^T -> concat-head AO[b][s][h*64+d]; r runs over consecutive d.
  float* aoq = AO + ((size_t)b * S_LEN + q0 + nl) * D_MODEL + h * D_K;
#pragma unroll
  for (int ds = 0; ds < 4; ++ds) {
    const int dbase = ds * 16 + 8 * half;
    float4 v0 = make_float4(acc[ds][0], acc[ds][1], acc[ds][2], acc[ds][3]);
    float4 v1 = make_float4(acc[ds][4], acc[ds][5], acc[ds][6], acc[ds][7]);
    *(float4*)(aoq + dbase)     = v0;
    *(float4*)(aoq + dbase + 4) = v1;
  }
}

// ---------------------------------------------------------------------------
// Kernel 3: output projection  Y = AO @ Wo^T + residual  (f32 out).
// ---------------------------------------------------------------------------
__global__ void outproj_kernel(const float* __restrict__ AO,
                               const float* __restrict__ W,
                               const float* __restrict__ resid,
                               float* __restrict__ Y) {
  const int lane = threadIdx.x & 31;
  const int wave = blockIdx.x * (blockDim.x >> 5) + (threadIdx.x >> 5);
  const int nt = wave & 63;
  const int mt = wave >> 6;
  const int m0 = mt * 64, n0 = nt * 16;

  v8f zero = {0.f,0.f,0.f,0.f,0.f,0.f,0.f,0.f};
  v8f acc[4] = {zero, zero, zero, zero};

  for (int k0 = 0; k0 < D_MODEL; k0 += 32) {
    v16h b = b_frag_f32(W, D_MODEL, n0, k0, lane);
#pragma unroll
    for (int i = 0; i < 4; ++i) {
      v16h a = a_frag_f32(AO, D_MODEL, m0 + 16 * i, k0, lane);
      acc[i] = wmma_f16(a, b, acc[i]);
    }
  }

  const int half = lane >> 4, nl = lane & 15;
  const int n = n0 + nl;
#pragma unroll
  for (int i = 0; i < 4; ++i) {
#pragma unroll
    for (int r = 0; r < 8; ++r) {
      const int m = m0 + i * 16 + r + 8 * half;
      const size_t idx = (size_t)m * D_MODEL + n;
      Y[idx] = acc[i][r] + resid[idx];
    }
  }
}

// ---------------------------------------------------------------------------
// Kernel 4: in-place LayerNorm over rows of 1024 (eps=1e-6, biased var).
// ---------------------------------------------------------------------------
__global__ void ln_kernel(float* __restrict__ Y,
                          const float* __restrict__ gamma,
                          const float* __restrict__ beta) {
  __shared__ float sb[8], ssb[8];
  const int lane = threadIdx.x & 31;
  const int w    = threadIdx.x >> 5;
  float* p = Y + (size_t)blockIdx.x * D_MODEL + threadIdx.x * 4;
  float4 x = *(const float4*)p;
  float s  = x.x + x.y + x.z + x.w;
  float ss = x.x*x.x + x.y*x.y + x.z*x.z + x.w*x.w;
#pragma unroll
  for (int o = 1; o < 32; o <<= 1) {
    s  += __shfl_xor(s,  o, 32);
    ss += __shfl_xor(ss, o, 32);
  }
  if (lane == 0) { sb[w] = s; ssb[w] = ss; }
  __syncthreads();
  float tot = 0.f, tot2 = 0.f;
#pragma unroll
  for (int i = 0; i < 8; ++i) { tot += sb[i]; tot2 += ssb[i]; }
  const float mu  = tot * (1.0f / D_MODEL);
  const float var = tot2 * (1.0f / D_MODEL) - mu * mu;
  const float inv = rsqrtf(var + 1e-6f);
  const float4 g = *(const float4*)(gamma + threadIdx.x * 4);
  const float4 be = *(const float4*)(beta + threadIdx.x * 4);
  float4 o;
  o.x = (x.x - mu) * inv * g.x + be.x;
  o.y = (x.y - mu) * inv * g.y + be.y;
  o.z = (x.z - mu) * inv * g.z + be.z;
  o.w = (x.w - mu) * inv * g.w + be.w;
  *(float4*)p = o;
}

// ---------------------------------------------------------------------------
extern "C" void kernel_launch(void* const* d_in, const int* in_sizes, int n_in,
                              void* d_out, int out_size, void* d_ws, size_t ws_size,
                              hipStream_t stream) {
  const float* q    = (const float*)d_in[0];
  const float* k    = (const float*)d_in[1];
  const float* v    = (const float*)d_in[2];
  const int*   mask = (const int*)  d_in[3];
  const float* wq   = (const float*)d_in[4];
  const float* wk   = (const float*)d_in[5];
  const float* wv   = (const float*)d_in[6];
  const float* wo   = (const float*)d_in[7];
  const float* lng  = (const float*)d_in[8];
  const float* lnb  = (const float*)d_in[9];

  // workspace layout (80 MB total):
  //   [0,16M)  Qh f16   [16M,32M) Kh f16   [32M,48M) VhT f16   [48M,80M) AO f32
  char* ws = (char*)d_ws;
  _Float16* Qh  = (_Float16*)(ws);
  _Float16* Kh  = (_Float16*)(ws + (16u << 20));
  _Float16* VhT = (_Float16*)(ws + (32u << 20));
  float*    AO  = (float*)   (ws + (48u << 20));

  float* out  = (float*)d_out;           // [4,2048,1024]
  float* attn = out + OUT_ELEMS;         // [4,16,2048,2048]

  dim3 blk(256);
  proj_kernel<0><<<dim3(1024), blk, 0, stream>>>(q, wq, Qh);
  proj_kernel<1><<<dim3(1024), blk, 0, stream>>>(k, wk, Kh);
  proj_kernel<2><<<dim3(1024), blk, 0, stream>>>(v, wv, VhT);
  attn_kernel   <<<dim3(1024), blk, 0, stream>>>(Qh, Kh, VhT, mask, attn, AO);
  outproj_kernel<<<dim3(1024), blk, 0, stream>>>(AO, wo, q, out);
  ln_kernel     <<<dim3(ROWS), blk, 0, stream>>>(out, lng, lnb);
}
